// MegaTransformerCausalSelfAttention_70858370449812
// MI455X (gfx1250) — matrix operand order repaired and checked by reading, not compiled, not run on hardware
//
#include <hip/hip_runtime.h>

// ---------------------------------------------------------------------------
// Problem constants (match the reference exactly)
// ---------------------------------------------------------------------------
#define D_MODEL 2048
#define SEQ     2048
#define BATCH   2
#define NH      32
#define NG      8
#define DH      64

// ---------------------------------------------------------------------------
// WMMA types / helpers (CDNA5 wave32, v_wmma_f32_16x16x32_bf16)
// ---------------------------------------------------------------------------
typedef __attribute__((ext_vector_type(16))) __bf16 bf16x16;
typedef __attribute__((ext_vector_type(8)))  float  f32x8;

union FragAB {           // one WMMA A/B operand: 16 bf16 = 8 VGPRs
  bf16x16 v;
  uint4   q[2];
};
union BF8 {              // 8 bf16 viewed as uint4
  uint4  q;
  __bf16 h[8];
};
union BF4 {              // 4 bf16 viewed as uint2 (packed LDS store)
  uint2  q;
  __bf16 h[4];
};

__device__ __forceinline__ f32x8 zero8() {
  f32x8 z;
#pragma unroll
  for (int i = 0; i < 8; ++i) z[i] = 0.0f;
  return z;
}

__device__ __forceinline__ f32x8 wmma_bf16(bf16x16 a, bf16x16 b, f32x8 c) {
  // (neg_a, A, neg_b, B, c_mod, C, reuse_a, reuse_b)
  return __builtin_amdgcn_wmma_f32_16x16x32_bf16(false, a, false, b, (short)0, c,
                                                 false, false);
}

// ---------------------------------------------------------------------------
// Async global -> LDS copy (GLOBAL_LOAD_ASYNC_TO_LDS_B128, ASYNCcnt tracked).
// Builtin signature (from hipcc diagnostic): pointers to 4xint GCC vectors,
// global source in AS(1), LDS destination in AS(3), imm offset, imm cpol.
// Guarded: falls back to a plain 16B copy if the builtin is unavailable.
// ---------------------------------------------------------------------------
#if defined(__has_builtin)
#if __has_builtin(__builtin_amdgcn_global_load_async_to_lds_b128)
#define HAVE_ASYNC_CP 1
#endif
#endif

typedef int v4i __attribute__((vector_size(4 * sizeof(int))));
typedef __attribute__((address_space(1))) v4i as1_v4i;
typedef __attribute__((address_space(3))) v4i as3_v4i;

__device__ __forceinline__ void cp_glds_b128(const void* g, void* l) {
#ifdef HAVE_ASYNC_CP
  __builtin_amdgcn_global_load_async_to_lds_b128((as1_v4i*)g, (as3_v4i*)l, 0, 0);
#else
  *(uint4*)l = *(const uint4*)g;
#endif
}

__device__ __forceinline__ void cp_glds_wait() {
#ifdef HAVE_ASYNC_CP
#if __has_builtin(__builtin_amdgcn_s_wait_asynccnt)
  __builtin_amdgcn_s_wait_asynccnt(0);
#else
  asm volatile("s_wait_asynccnt 0" ::: "memory");
#endif
#endif
}

// ---------------------------------------------------------------------------
// Generic fp32-in GEMM via bf16 WMMA:  C(M,N) = A(M,K) * B(K,N)
//   block tile 128x128, K-step 64, 256 threads = 8 waves, wave tile 64x32.
// MODE 0: scatter bf16 output to (batch, heads, SEQ, DH)  [QKV projections]
// MODE 1: fp32 output row-major with bias                 [final projection]
// All dims are multiples of the tile sizes for this problem -> no guards.
// ---------------------------------------------------------------------------
#define G_TM   128
#define G_TN   128
#define G_TK   64
#define G_STR  72           // LDS row stride in bf16 (64 + 8 pad), 16B aligned

template <int MODE>
__global__ __launch_bounds__(256) void gemm_kernel(
    const float* __restrict__ A, const float* __restrict__ B,
    void* __restrict__ Out, const float* __restrict__ bias,
    int M, int N, int K, int heads) {
  __shared__ __bf16 sA[G_TM * G_STR];   // A tile, row-major (m, k)
  __shared__ __bf16 sBt[G_TN * G_STR];  // B tile transposed (n, k)

  const int tid  = threadIdx.x;
  const int lane = tid & 31;
  const int wave = tid >> 5;
  const int half = lane >> 4;   // 0: lanes 0-15, 1: lanes 16-31
  const int l16  = lane & 15;

  const int bm = blockIdx.y * G_TM;
  const int bn = blockIdx.x * G_TN;
  const int wy = (wave >> 2) * 64;  // wave M offset within block (0/64)
  const int wx = (wave & 3) * 32;   // wave N offset within block (0/32/64/96)

  f32x8 acc[4][2];
#pragma unroll
  for (int i = 0; i < 4; ++i)
#pragma unroll
    for (int j = 0; j < 2; ++j) acc[i][j] = zero8();

  for (int k0 = 0; k0 < K; k0 += G_TK) {
    // ---- prefetch next K tile (global_prefetch_b8 hints) ----
    if (k0 + G_TK < K) {
      __builtin_prefetch(A + (size_t)(bm + (tid >> 1)) * K + k0 + G_TK + (tid & 1) * 32, 0, 3);
      __builtin_prefetch(B + (size_t)(k0 + G_TK + (tid >> 2)) * N + bn + (tid & 3) * 32, 0, 3);
    }
    // ---- stage A tile: 128x64 fp32 -> bf16, packed ds_store_b64 ----
#pragma unroll
    for (int it = 0; it < 8; ++it) {
      int c  = it * 256 + tid;          // 0..2047 float4 chunks
      int mr = c >> 4;                  // row in tile
      int kc = (c & 15) * 4;            // col in tile
      float4 f = *(const float4*)(A + (size_t)(bm + mr) * K + k0 + kc);
      BF4 u;
      u.h[0] = (__bf16)f.x; u.h[1] = (__bf16)f.y;
      u.h[2] = (__bf16)f.z; u.h[3] = (__bf16)f.w;
      *(uint2*)(sA + mr * G_STR + kc) = u.q;
    }
    // ---- stage B tile transposed: 64x128 fp32 -> bf16 (n,k) scatter ----
#pragma unroll
    for (int it = 0; it < 8; ++it) {
      int c  = it * 256 + tid;
      int kr = c >> 5;                  // k row in tile
      int nc = (c & 31) * 4;            // n col in tile
      float4 f = *(const float4*)(B + (size_t)(k0 + kr) * N + bn + nc);
      sBt[(nc + 0) * G_STR + kr] = (__bf16)f.x;
      sBt[(nc + 1) * G_STR + kr] = (__bf16)f.y;
      sBt[(nc + 2) * G_STR + kr] = (__bf16)f.z;
      sBt[(nc + 3) * G_STR + kr] = (__bf16)f.w;
    }
    __syncthreads();

    // ---- two K=32 WMMA steps per staged tile ----
#pragma unroll
    for (int kk = 0; kk < G_TK; kk += 32) {
      FragAB fa[4], fb[2];
#pragma unroll
      for (int i = 0; i < 4; ++i) {     // A frags: rows wy + i*16
        const __bf16* p = sA + (wy + i * 16 + l16) * G_STR + kk + half * 8;
        fa[i].q[0] = *(const uint4*)(p);
        fa[i].q[1] = *(const uint4*)(p + 16);
      }
#pragma unroll
      for (int j = 0; j < 2; ++j) {     // B frags: cols wx + j*16
        const __bf16* p = sBt + (wx + j * 16 + l16) * G_STR + kk + half * 8;
        fb[j].q[0] = *(const uint4*)(p);
        fb[j].q[1] = *(const uint4*)(p + 16);
      }
#pragma unroll
      for (int i = 0; i < 4; ++i)
#pragma unroll
        for (int j = 0; j < 2; ++j)
          acc[i][j] = wmma_bf16(fa[i].v, fb[j].v, acc[i][j]);
    }
    __syncthreads();
  }

  // ---- epilogue: C layout row = v + 8*half, col = l16 ----
#pragma unroll
  for (int i = 0; i < 4; ++i)
#pragma unroll
    for (int j = 0; j < 2; ++j)
#pragma unroll
      for (int v = 0; v < 8; ++v) {
        int row = bm + wy + i * 16 + v + half * 8;
        int col = bn + wx + j * 16 + l16;
        float val = acc[i][j][v];
        if (MODE == 0) {
          __bf16* O = (__bf16*)Out;
          int bb = row >> 11, ss = row & (SEQ - 1);   // SEQ == 2048
          int hh = col >> 6, dd = col & 63;
          O[(((size_t)bb * heads + hh) * SEQ + ss) * DH + dd] = (__bf16)val;
        } else {
          float* O = (float*)Out;
          O[(size_t)row * N + col] = val + bias[col];
        }
      }
}

// ---------------------------------------------------------------------------
// Interleaved RoPE on a bf16 (BH, S, D) tensor, one thread per pair
// ---------------------------------------------------------------------------
__global__ __launch_bounds__(256) void rope_kernel(__bf16* X, int BH) {
  int idx = blockIdx.x * blockDim.x + threadIdx.x;
  int total = BH * SEQ * (DH / 2);
  if (idx >= total) return;
  int p   = idx % (DH / 2);
  int rem = idx / (DH / 2);
  int s   = rem & (SEQ - 1);
  int bh  = rem >> 11;
  float inv = __expf(-__logf(10000.0f) * (2.0f * (float)p) / (float)DH);
  float ang = (float)s * inv;
  float cs = __cosf(ang), sn = __sinf(ang);
  __bf16* ptr = X + ((size_t)bh * SEQ + s) * DH + 2 * p;
  float x0 = (float)ptr[0], x1 = (float)ptr[1];
  ptr[0] = (__bf16)(x0 * cs - x1 * sn);
  ptr[1] = (__bf16)(x1 * cs + x0 * sn);
}

// ---------------------------------------------------------------------------
// Flash attention: one workgroup per (row-block of 128, head, batch).
//   8 waves, each owns a 16-row WMMA tile. Online softmax with ALiBi,
//   grok 30*tanh(s/30) rescale, causal mask. GQA: head h -> group h/4.
//   K tile staged via async global->LDS copies (ASYNCcnt).
// ---------------------------------------------------------------------------
#define A_BM   128
#define A_BN   128
#define K_STR  72     // K tile LDS stride (64 + 8)
#define V_STR  136    // Vt tile LDS stride (128 + 8)
#define P_STR  136    // per-wave P strip stride

__global__ __launch_bounds__(256) void attn_kernel(
    const __bf16* __restrict__ Q, const __bf16* __restrict__ Kg,
    const __bf16* __restrict__ Vg, float* __restrict__ Ctx) {
  __shared__ __bf16 sK[A_BN * K_STR];       // (t, d) : rows directly = B frags
  __shared__ __bf16 sVt[DH * V_STR];        // (d, t) : transposed V
  __shared__ __bf16 sP[8 * 16 * P_STR];     // per-wave 16x128 P strip

  const int tid  = threadIdx.x;
  const int lane = tid & 31;
  const int wave = tid >> 5;
  const int half = lane >> 4;
  const int l16  = lane & 15;

  const int rb = blockIdx.x;                // row block
  const int h  = blockIdx.y;                // head
  const int b  = blockIdx.z;                // batch
  const int g  = h >> 2;                    // GQA group (32/8 = 4 rep)
  const size_t qbase = ((size_t)(b * NH + h)) * SEQ * DH;
  const size_t kbase = ((size_t)(b * NG + g)) * SEQ * DH;

  const int row0 = rb * A_BM + wave * 16;   // this wave's 16 query rows

  // ---- preload Q A-fragments (D=64 -> two K=32 chunks), reused all tiles ----
  FragAB qf[2];
  {
    const __bf16* qr = Q + qbase + (size_t)(row0 + l16) * DH;
#pragma unroll
    for (int c = 0; c < 2; ++c) {
      qf[c].q[0] = *(const uint4*)(qr + c * 32 + half * 8);
      qf[c].q[1] = *(const uint4*)(qr + c * 32 + 16 + half * 8);
    }
  }

  float mrow[8], lrow[8];
  f32x8 oacc[4];
#pragma unroll
  for (int v = 0; v < 8; ++v) { mrow[v] = -1e30f; lrow[v] = 0.0f; }
#pragma unroll
  for (int j = 0; j < 4; ++j) oacc[j] = zero8();

  const float slope = exp2f(-0.25f * (float)(h + 1));  // 2^(-(h+1)*8/32)
  const int ntiles = rb + 1;                           // causal: tiles <= diag

  for (int t = 0; t < ntiles; ++t) {
    const int kt0 = t * A_BN;
    // ---- stage K tile (128 x 64 bf16) via async global->LDS b128 copies ----
#pragma unroll
    for (int it = 0; it < 4; ++it) {
      int c  = it * 256 + tid;      // chunk-of-8
      int tr = c >> 3;
      int dc = (c & 7) * 8;
      cp_glds_b128(Kg + kbase + (size_t)(kt0 + tr) * DH + dc,
                   sK + tr * K_STR + dc);
    }
    // ---- stage V tile transposed (d, t) ----
#pragma unroll
    for (int it = 0; it < 4; ++it) {
      int c  = it * 256 + tid;
      int tr = c >> 3;
      int dc = (c & 7) * 8;
      BF8 u;
      u.q = *(const uint4*)(Vg + kbase + (size_t)(kt0 + tr) * DH + dc);
#pragma unroll
      for (int j = 0; j < 8; ++j) sVt[(dc + j) * V_STR + tr] = u.h[j];
    }
    cp_glds_wait();                 // ASYNCcnt -> 0 for this wave's copies
    __syncthreads();

    // ---- scores: 8 column tiles x 2 K-steps of WMMA ----
    f32x8 sc[8];
#pragma unroll
    for (int n = 0; n < 8; ++n) {
      f32x8 a = zero8();
#pragma unroll
      for (int c = 0; c < 2; ++c) {
        FragAB kb;  // B frag = row (key t) of K, contiguous d
        const __bf16* p = sK + (n * 16 + l16) * K_STR + c * 32 + half * 8;
        kb.q[0] = *(const uint4*)(p);
        kb.q[1] = *(const uint4*)(p + 16);
        a = wmma_bf16(qf[c].v, kb.v, a);
      }
      sc[n] = a;
    }

    // ---- ALiBi + scale + 30*tanh(s/30) + causal mask ----
#pragma unroll
    for (int n = 0; n < 8; ++n)
#pragma unroll
      for (int v = 0; v < 8; ++v) {
        int r = row0 + v + half * 8;
        int cc = kt0 + n * 16 + l16;
        float s = sc[n][v];
        s = (s - slope * fabsf((float)(r - cc))) * 0.125f;  // bias then /sqrt(64)
        s = 30.0f * tanhf(s * (1.0f / 30.0f));
        if (cc > r) s = -1e30f;                             // causal
        sc[n][v] = s;
      }

    // ---- online softmax: row max / exp / row sum ----
    float tmax[8];
#pragma unroll
    for (int v = 0; v < 8; ++v) {
      float m = sc[0][v];
#pragma unroll
      for (int n = 1; n < 8; ++n) m = fmaxf(m, sc[n][v]);
      tmax[v] = m;
    }
#pragma unroll
    for (int off = 1; off < 16; off <<= 1)
#pragma unroll
      for (int v = 0; v < 8; ++v)
        tmax[v] = fmaxf(tmax[v], __shfl_xor(tmax[v], off, 32));

    float scale[8], tsum[8];
#pragma unroll
    for (int v = 0; v < 8; ++v) {
      float mn = fmaxf(mrow[v], tmax[v]);
      scale[v] = __expf(mrow[v] - mn);
      mrow[v] = mn;
      tsum[v] = 0.0f;
    }
#pragma unroll
    for (int n = 0; n < 8; ++n)
#pragma unroll
      for (int v = 0; v < 8; ++v) {
        float p = __expf(sc[n][v] - mrow[v]);
        sc[n][v] = p;
        tsum[v] += p;
      }
#pragma unroll
    for (int off = 1; off < 16; off <<= 1)
#pragma unroll
      for (int v = 0; v < 8; ++v) tsum[v] += __shfl_xor(tsum[v], off, 32);
#pragma unroll
    for (int v = 0; v < 8; ++v) lrow[v] = lrow[v] * scale[v] + tsum[v];
#pragma unroll
    for (int j = 0; j < 4; ++j)
#pragma unroll
      for (int v = 0; v < 8; ++v) oacc[j][v] *= scale[v];

    // ---- P: C layout -> A layout via per-wave LDS strip ----
    __bf16* pw = sP + wave * 16 * P_STR;
#pragma unroll
    for (int n = 0; n < 8; ++n)
#pragma unroll
      for (int v = 0; v < 8; ++v)
        pw[(v + half * 8) * P_STR + n * 16 + l16] = (__bf16)sc[n][v];
    asm volatile("s_wait_dscnt 0" ::: "memory");  // wave-private LDS RAW

    // ---- ctx += P (16x128) * V (128x64): 4 K-steps x 4 d-tiles ----
#pragma unroll
    for (int c = 0; c < 4; ++c) {
      FragAB pf;
      const __bf16* pp = pw + l16 * P_STR + c * 32 + half * 8;
      pf.q[0] = *(const uint4*)(pp);
      pf.q[1] = *(const uint4*)(pp + 16);
#pragma unroll
      for (int j = 0; j < 4; ++j) {
        FragAB vb;  // B frag = row d of Vt, contiguous t
        const __bf16* vp = sVt + (j * 16 + l16) * V_STR + c * 32 + half * 8;
        vb.q[0] = *(const uint4*)(vp);
        vb.q[1] = *(const uint4*)(vp + 16);
        oacc[j] = wmma_bf16(pf.v, vb.v, oacc[j]);
      }
    }
    __syncthreads();
  }

  // ---- normalize and store ctx as (B, S, H*DH) fp32 ----
#pragma unroll
  for (int j = 0; j < 4; ++j)
#pragma unroll
    for (int v = 0; v < 8; ++v) {
      int r = row0 + v + half * 8;
      int d = j * 16 + l16;
      Ctx[((size_t)b * SEQ + r) * (NH * DH) + h * DH + d] = oacc[j][v] / lrow[v];
    }
}

// ---------------------------------------------------------------------------
// Launcher
// ---------------------------------------------------------------------------
extern "C" void kernel_launch(void* const* d_in, const int* in_sizes, int n_in,
                              void* d_out, int out_size, void* d_ws, size_t ws_size,
                              hipStream_t stream) {
  const float* X  = (const float*)d_in[0];  // (2, 2048, 2048)
  const float* Wq = (const float*)d_in[1];  // (2048, 2048)
  const float* Wk = (const float*)d_in[2];  // (2048, 512)
  const float* Wv = (const float*)d_in[3];  // (2048, 512)
  const float* Wo = (const float*)d_in[4];  // (2048, 2048)
  const float* bo = (const float*)d_in[5];  // (2048,)
  float* out = (float*)d_out;

  char* ws = (char*)d_ws;
  __bf16* Qw = (__bf16*)(ws);                            // 16 MB  (B,NH,S,DH)
  __bf16* Kw = (__bf16*)(ws + (size_t)16 * 1024 * 1024); //  4 MB  (B,NG,S,DH)
  __bf16* Vw = (__bf16*)(ws + (size_t)20 * 1024 * 1024); //  4 MB  (B,NG,S,DH)
  float*  Cw = (float*) (ws + (size_t)24 * 1024 * 1024); // 32 MB  (B,S,NH*DH)

  const int M = BATCH * SEQ;  // 4096

  // QKV projections (bf16 WMMA GEMMs with head-scatter epilogue)
  gemm_kernel<0><<<dim3(D_MODEL / G_TN, M / G_TM), 256, 0, stream>>>(
      X, Wq, Qw, nullptr, M, NH * DH, D_MODEL, NH);
  gemm_kernel<0><<<dim3((NG * DH) / G_TN, M / G_TM), 256, 0, stream>>>(
      X, Wk, Kw, nullptr, M, NG * DH, D_MODEL, NG);
  gemm_kernel<0><<<dim3((NG * DH) / G_TN, M / G_TM), 256, 0, stream>>>(
      X, Wv, Vw, nullptr, M, NG * DH, D_MODEL, NG);

  // RoPE on Q and K
  {
    int totQ = BATCH * NH * SEQ * (DH / 2);
    int totK = BATCH * NG * SEQ * (DH / 2);
    rope_kernel<<<(totQ + 255) / 256, 256, 0, stream>>>(Qw, BATCH * NH);
    rope_kernel<<<(totK + 255) / 256, 256, 0, stream>>>(Kw, BATCH * NG);
  }

  // Flash attention
  attn_kernel<<<dim3(SEQ / A_BM, NH, BATCH), 256, 0, stream>>>(Qw, Kw, Vw, Cw);

  // Output projection with bias
  gemm_kernel<1><<<dim3(D_MODEL / G_TN, M / G_TM), 256, 0, stream>>>(
      Cw, Wo, out, bo, M, D_MODEL, D_MODEL, NH);
}
